// Decoder_73065983639912
// MI455X (gfx1250) — matrix operand order repaired and checked by reading, not compiled, not run on hardware
//
#include <hip/hip_runtime.h>
#include <hip/hip_bf16.h>

// ---------------------------------------------------------------------------
// Problem constants (from reference)
// ---------------------------------------------------------------------------
#define BB   1024
#define SS   200
#define EE   128
#define HH   8
#define DH   16
#define K3E  384          // 3*E
#define NEGV (-1e9f)
#define CSCALE 10.0f

typedef __attribute__((ext_vector_type(16))) _Float16 v16h;
typedef __attribute__((ext_vector_type(8)))  float    v8f;

// ---------------------------------------------------------------------------
// Generic GEMM: D[M,N] = A_f32[M,K(lda)] x B (given as Bt_f16[N][K], ldb=K)
// wave32, 1 wave per 16x16 output tile, 4 waves/block cover 64 columns.
// Uses V_WMMA_F32_16X16X32_F16 (A converted f32->f16 in-register).
// Fragment K order: contiguous, k = 16*half + s, same for A and B.
// ---------------------------------------------------------------------------
template <bool STORE_F16>
__global__ __launch_bounds__(128) void gemm_a32_bt16(
    const float* __restrict__ A, const _Float16* __restrict__ Bt,
    void* __restrict__ Dptr, int N, int K, int lda)
{
    const int lane  = threadIdx.x & 31;
    const int wave  = threadIdx.x >> 5;
    const int mtile = blockIdx.x;
    const int ntile = blockIdx.y * 4 + wave;
    const int hsel  = lane >> 4;        // which half of the wave
    const int l15   = lane & 15;

    const int arow  = mtile * 16 + l15;
    const int kA    = hsel * 16;        // A frag: K base {0,16} within 32-chunk
    const int kB    = hsel * 16;        // B frag: K base {0,16} within 32-chunk
    const int ncol  = ntile * 16 + l15;

    const float*    ap = A  + (size_t)arow * lda;
    const _Float16* bp = Bt + (size_t)ncol * K;

    v8f acc = {};
    for (int kc = 0; kc < K; kc += 32) {
        // ---- A fragment: rows of A, f32 -> f16 ----
        const float4 f0 = *(const float4*)(ap + kc + kA);
        const float4 f1 = *(const float4*)(ap + kc + kA + 4);
        const float4 f2 = *(const float4*)(ap + kc + kA + 8);
        const float4 f3 = *(const float4*)(ap + kc + kA + 12);
        v16h a;
        a[0]=(_Float16)f0.x;  a[1]=(_Float16)f0.y;  a[2]=(_Float16)f0.z;  a[3]=(_Float16)f0.w;
        a[4]=(_Float16)f1.x;  a[5]=(_Float16)f1.y;  a[6]=(_Float16)f1.z;  a[7]=(_Float16)f1.w;
        a[8]=(_Float16)f2.x;  a[9]=(_Float16)f2.y;  a[10]=(_Float16)f2.z; a[11]=(_Float16)f2.w;
        a[12]=(_Float16)f3.x; a[13]=(_Float16)f3.y; a[14]=(_Float16)f3.z; a[15]=(_Float16)f3.w;
        // ---- B fragment: 16 consecutive K at column ncol (Bt row-major) ----
        v16h b = *(const v16h*)(bp + kc + kB);
        acc = __builtin_amdgcn_wmma_f32_16x16x32_f16(
                  false, a, false, b, (short)0, acc, false, false);
    }

    // ---- store D: VGPR r -> row (mtile*16 + hsel*8 + r), col ncol ----
    const int mbase = mtile * 16 + hsel * 8;
    if (STORE_F16) {
        _Float16* D = (_Float16*)Dptr;
        #pragma unroll
        for (int r = 0; r < 8; ++r)
            D[(size_t)(mbase + r) * N + ncol] = (_Float16)acc[r];
    } else {
        float* D = (float*)Dptr;
        #pragma unroll
        for (int r = 0; r < 8; ++r)
            D[(size_t)(mbase + r) * N + ncol] = acc[r];
    }
}

// ---------------------------------------------------------------------------
// Init: transpose + f16-convert weights. Wqt[128][384], W*t[128][128].
// ---------------------------------------------------------------------------
__global__ void prep_weights(const float* __restrict__ Wq, const float* __restrict__ Wk,
                             const float* __restrict__ Wv, const float* __restrict__ Wo,
                             const float* __restrict__ Wp,
                             _Float16* Wqt, _Float16* Wkt, _Float16* Wvt,
                             _Float16* Wot, _Float16* Wpt)
{
    int idx = blockIdx.x * 256 + threadIdx.x;
    if (idx < K3E * EE) {                     // Wq: [384,128] -> [128][384]
        int k = idx / EE, n = idx % EE;
        Wqt[n * K3E + k] = (_Float16)Wq[idx];
    }
    if (idx < EE * EE) {                      // [128,128] -> [128][128]^T
        int k = idx / EE, n = idx % EE;
        Wkt[n * EE + k] = (_Float16)Wk[idx];
        Wvt[n * EE + k] = (_Float16)Wv[idx];
        Wot[n * EE + k] = (_Float16)Wo[idx];
        Wpt[n * EE + k] = (_Float16)Wp[idx];
    }
}

// Init: hc[b] = [mean_s enc, vl, vf]
__global__ __launch_bounds__(128) void init_hc(const float* __restrict__ enc,
                                               const float* __restrict__ vl,
                                               const float* __restrict__ vf,
                                               float* __restrict__ hc)
{
    const int b = blockIdx.x, e = threadIdx.x;
    const float* p = enc + (size_t)b * SS * EE + e;
    float acc = 0.f;
    for (int s = 0; s < SS; ++s) acc += p[(size_t)s * EE];
    hc[b * K3E + e]           = acc * (1.0f / (float)SS);
    hc[b * K3E + EE + e]      = vl[e];
    hc[b * K3E + 2 * EE + e]  = vf[e];
}

__global__ void init_state(unsigned char* __restrict__ mask, float* __restrict__ logp)
{
    int idx = blockIdx.x * 256 + threadIdx.x;
    if (idx < BB * SS) mask[idx] = 0;
    if (idx < BB)      logp[idx] = 0.f;
}

// ---------------------------------------------------------------------------
// Per-step attention: scores -> masked softmax -> ctx.  1 block per b.
// ---------------------------------------------------------------------------
__global__ __launch_bounds__(256) void attn_step(
    const float* __restrict__ q,                // [B,128]
    const _Float16* __restrict__ kh,            // [B,S,128]
    const _Float16* __restrict__ vh,            // [B,S,128]
    const unsigned char* __restrict__ mask,     // [B,S]
    float* __restrict__ ctx)                    // [B,128]
{
    const int b = blockIdx.x, tid = threadIdx.x;
    __shared__ float sq[EE];
    __shared__ float sattn[HH][SS + 8];
    __shared__ float srowsum[HH];

    if (tid < EE) sq[tid] = q[b * EE + tid];
    __syncthreads();

    // scores[h][s] = (q_h . kh[b,s,h]) / sqrt(dh); mask -> NEG
    for (int idx = tid; idx < HH * SS; idx += 256) {
        const int h = idx / SS, s = idx - h * SS;
        const _Float16* kp = kh + ((size_t)b * SS + s) * EE + h * DH;
        float acc = 0.f;
        #pragma unroll
        for (int d = 0; d < DH; ++d) acc += sq[h * DH + d] * (float)kp[d];
        acc *= 0.25f;                           // 1/sqrt(16)
        if (mask[b * SS + s]) acc = NEGV;
        sattn[h][s] = acc;
    }
    __syncthreads();

    // softmax row per wave (8 waves <-> 8 heads)
    const int w = tid >> 5, lane = tid & 31;
    {
        float m = -3.0e38f;
        for (int s = lane; s < SS; s += 32) m = fmaxf(m, sattn[w][s]);
        #pragma unroll
        for (int o = 16; o; o >>= 1) m = fmaxf(m, __shfl_xor(m, o, 32));
        float sum = 0.f;
        for (int s = lane; s < SS; s += 32) {
            float e = __expf(sattn[w][s] - m);
            sattn[w][s] = e;
            sum += e;
        }
        #pragma unroll
        for (int o = 16; o; o >>= 1) sum += __shfl_xor(sum, o, 32);
        if (lane == 0) srowsum[w] = sum;
    }
    __syncthreads();

    // ctx[b, h*16+d] = sum_s attn * vh / rowsum
    if (tid < EE) {
        const int h = tid >> 4;
        const _Float16* vp = vh + (size_t)b * SS * EE + tid;
        float acc = 0.f;
        for (int s = 0; s < SS; ++s) acc += sattn[h][s] * (float)vp[(size_t)s * EE];
        ctx[b * EE + tid] = acc / srowsum[h];
    }
}

// ---------------------------------------------------------------------------
// Per-step pointer logits + greedy decode + state update.  1 block per b.
// ---------------------------------------------------------------------------
__global__ __launch_bounds__(256) void pointer_step(
    const float* __restrict__ o,                // [B,128] = ctx @ Wo
    const _Float16* __restrict__ kp16,          // [B,S,128] = enc @ Wproj
    const float* __restrict__ enc,              // [B,S,128]
    unsigned char* __restrict__ mask,           // [B,S]
    float* __restrict__ hc,                     // [B,384]
    float* __restrict__ out_logits,             // [B,S,S]
    float* __restrict__ out_logp,               // [B]
    float* __restrict__ out_sol,                // [B,S]
    int t)
{
    const int b = blockIdx.x, tid = threadIdx.x;
    const int w = tid >> 5, lane = tid & 31;
    __shared__ float so[EE];
    __shared__ float su[SS];                    // raw logits u
    __shared__ float sum_m[SS];                 // masked u
    __shared__ float red[256];
    __shared__ int   redi[256];

    if (tid < EE) so[tid] = o[b * EE + tid];
    __syncthreads();

    const float inv_scale_u = 0.08838834764831845f;   // 1/sqrt(128)
    // one wave per s: 32-lane dot(o, kproj[b,s]) then shuffle-reduce
    for (int s = w; s < SS; s += 8) {
        const _Float16* kp = kp16 + ((size_t)b * SS + s) * EE + lane * 4;
        float acc = so[lane * 4 + 0] * (float)kp[0]
                  + so[lane * 4 + 1] * (float)kp[1]
                  + so[lane * 4 + 2] * (float)kp[2]
                  + so[lane * 4 + 3] * (float)kp[3];
        #pragma unroll
        for (int ofs = 16; ofs; ofs >>= 1) acc += __shfl_xor(acc, ofs, 32);
        if (lane == 0) {
            float u = tanhf(acc * inv_scale_u) * CSCALE;
            su[s] = u;
            sum_m[s] = mask[b * SS + s] ? NEGV : u;
            out_logits[((size_t)b * SS + t) * SS + s] = u;   // raw (unmasked) logits
        }
    }
    __syncthreads();

    // argmax over masked logits (first-occurrence tie-break like jnp.argmax)
    float bv = -3.0e38f; int bi = 0;
    for (int s = tid; s < SS; s += 256) { float v = sum_m[s]; if (v > bv) { bv = v; bi = s; } }
    red[tid] = bv; redi[tid] = bi;
    __syncthreads();
    for (int stride = 128; stride; stride >>= 1) {
        if (tid < stride) {
            float rv = red[tid + stride]; int ri = redi[tid + stride];
            if (rv > red[tid] || (rv == red[tid] && ri < redi[tid])) { red[tid] = rv; redi[tid] = ri; }
        }
        __syncthreads();
    }
    const int   city = redi[0];
    const float maxv = red[0];
    __syncthreads();

    // log-sum-exp of masked logits
    float ps = 0.f;
    for (int s = tid; s < SS; s += 256) ps += __expf(sum_m[s] - maxv);
    red[tid] = ps;
    __syncthreads();
    for (int stride = 128; stride; stride >>= 1) {
        if (tid < stride) red[tid] += red[tid + stride];
        __syncthreads();
    }

    if (tid == 0) {
        float lse = logf(red[0]) + maxv;
        out_logp[b] += lse - su[city];           // -= log_softmax(um)[city]
        out_sol[b * SS + t] = (float)city;
        mask[b * SS + city] = 1;
    }
    // update hc: last = enc[b,city]; first = last when t==0
    if (tid < EE) {
        float v = enc[((size_t)b * SS + city) * EE + tid];
        hc[b * K3E + EE + tid] = v;
        if (t == 0) hc[b * K3E + 2 * EE + tid] = v;
    }
}

// ---------------------------------------------------------------------------
// Host orchestration
// ---------------------------------------------------------------------------
static inline size_t align256(size_t x) { return (x + 255) & ~(size_t)255; }

extern "C" void kernel_launch(void* const* d_in, const int* in_sizes, int n_in,
                              void* d_out, int out_size, void* d_ws, size_t ws_size,
                              hipStream_t stream)
{
    const float* enc = (const float*)d_in[0];   // [B,S,E]
    const float* vl  = (const float*)d_in[1];   // [E]
    const float* vf  = (const float*)d_in[2];   // [E]
    const float* Wq  = (const float*)d_in[3];   // [384,128]
    const float* Wk  = (const float*)d_in[4];   // [128,128]
    const float* Wv  = (const float*)d_in[5];
    const float* Wo  = (const float*)d_in[6];
    const float* Wp  = (const float*)d_in[7];

    float* out_logits = (float*)d_out;                          // [B,S,S]
    float* out_logp   = out_logits + (size_t)BB * SS * SS;      // [B]
    float* out_sol    = out_logp + BB;                          // [B,S]

    // workspace carve-up
    char* ws = (char*)d_ws;
    size_t off = 0;
    const size_t szBSE16 = (size_t)BB * SS * EE * sizeof(_Float16);
    _Float16* kh16 = (_Float16*)(ws + off); off = align256(off + szBSE16);
    _Float16* vh16 = (_Float16*)(ws + off); off = align256(off + szBSE16);
    _Float16* kp16 = (_Float16*)(ws + off); off = align256(off + szBSE16);
    _Float16* Wqt  = (_Float16*)(ws + off); off = align256(off + (size_t)EE * K3E * 2);
    _Float16* Wkt  = (_Float16*)(ws + off); off = align256(off + (size_t)EE * EE * 2);
    _Float16* Wvt  = (_Float16*)(ws + off); off = align256(off + (size_t)EE * EE * 2);
    _Float16* Wot  = (_Float16*)(ws + off); off = align256(off + (size_t)EE * EE * 2);
    _Float16* Wpt  = (_Float16*)(ws + off); off = align256(off + (size_t)EE * EE * 2);
    float* hc      = (float*)(ws + off);    off = align256(off + (size_t)BB * K3E * 4);
    float* qbuf    = (float*)(ws + off);    off = align256(off + (size_t)BB * EE * 4);
    float* ctxbuf  = (float*)(ws + off);    off = align256(off + (size_t)BB * EE * 4);
    float* obuf    = (float*)(ws + off);    off = align256(off + (size_t)BB * EE * 4);
    unsigned char* mask = (unsigned char*)(ws + off); off = align256(off + (size_t)BB * SS);

    // ---- one-time prep ----
    prep_weights<<<(K3E * EE + 255) / 256, 256, 0, stream>>>(
        Wq, Wk, Wv, Wo, Wp, Wqt, Wkt, Wvt, Wot, Wpt);
    init_hc<<<BB, EE, 0, stream>>>(enc, vl, vf, hc);
    init_state<<<(BB * SS + 255) / 256, 256, 0, stream>>>(mask, out_logp);

    // ---- precompute kh / vh / kproj via WMMA GEMMs (M = B*S) ----
    {
        dim3 grid((BB * SS) / 16, EE / 64);
        gemm_a32_bt16<true><<<grid, 128, 0, stream>>>(enc, Wkt, (void*)kh16, EE, EE, EE);
        gemm_a32_bt16<true><<<grid, 128, 0, stream>>>(enc, Wvt, (void*)vh16, EE, EE, EE);
        gemm_a32_bt16<true><<<grid, 128, 0, stream>>>(enc, Wpt, (void*)kp16, EE, EE, EE);
    }

    // ---- sequential decode ----
    dim3 gridQ(BB / 16, EE / 64);
    for (int t = 0; t < SS; ++t) {
        // q = hc @ Wq  (K = 384)
        gemm_a32_bt16<false><<<gridQ, 128, 0, stream>>>(hc, Wqt, (void*)qbuf, EE, K3E, K3E);
        // attention -> ctx
        attn_step<<<BB, 256, 0, stream>>>(qbuf, kh16, vh16, mask, ctxbuf);
        // o = ctx @ Wo
        gemm_a32_bt16<false><<<gridQ, 128, 0, stream>>>(ctxbuf, Wot, (void*)obuf, EE, EE, EE);
        // pointer logits, argmax, logp, state update
        pointer_step<<<BB, 256, 0, stream>>>(obuf, kp16, enc, mask, hc,
                                             out_logits, out_logp, out_sol, t);
    }
}
